// RNNClassifier_6871947673899
// MI455X (gfx1250) — matrix operand order repaired and checked by reading, not compiled, not run on hardware
//
#include <hip/hip_runtime.h>
#include <hip/hip_bf16.h>

// ---------------------------------------------------------------------------
// RNNClassifier for MI455X (gfx1250, wave32, WMMA)
// B=256, T=128, I=512, H=512, heads=8, hd=64, C=2
// Attention is per-timestep over the batch axis and the output is sliced to
// t=T-1, so only the final hidden state h_T feeds qkv/attention/out_proj/fc.
// ---------------------------------------------------------------------------

typedef __attribute__((ext_vector_type(16))) _Float16 v16h;
typedef __attribute__((ext_vector_type(8)))  _Float16 v8h;
typedef __attribute__((ext_vector_type(8)))  float    v8f;

union F16Frag { v16h v; v8h h8[2]; };

// A-matrix 16x32 f16 fragment (row-major source, leading dim ld).
// Lane l: row = l&15 ; lanes 0-15 hold K 0..7 & 16..23, lanes 16-31 hold 8..15 & 24..31.
__device__ __forceinline__ v16h load_fragA(const _Float16* __restrict__ p, int ld) {
  const int lane = threadIdx.x & 31;
  const int r    = lane & 15;
  const int kc   = (lane >> 4) << 3;   // 0 or 8
  const _Float16* q = p + (size_t)r * ld + kc;
  F16Frag f;
  f.h8[0] = *(const v8h*)(q);
  f.h8[1] = *(const v8h*)(q + 16);
  return f.v;
}

// B-matrix 32x16 f16 fragment for D = A @ W^T (W row-major [N x K], leading dim ld).
// Lane l holds column n = l&15 of B, i.e. row n of W; lanes 0-15 K 0..7 & 16..23, lanes 16-31 K 8..15 & 24..31.
__device__ __forceinline__ v16h load_fragB(const _Float16* __restrict__ p, int ld) {
  const int lane = threadIdx.x & 31;
  const int c    = lane & 15;
  const int kc   = (lane >> 4) << 3;   // 0 or 8
  const _Float16* q = p + (size_t)c * ld + kc;
  F16Frag f;
  f.h8[0] = *(const v8h*)(q);
  f.h8[1] = *(const v8h*)(q + 16);
  return f.v;
}

__device__ __forceinline__ v8f wmma_f16(v16h a, v16h b, v8f c) {
  return __builtin_amdgcn_wmma_f32_16x16x32_f16(false, a, false, b, (short)0, c, false, false);
}

// ---------------------------------------------------------------------------
// f32 -> f16 conversion (grid-stride)
// ---------------------------------------------------------------------------
__global__ void k_f32_to_f16(const float* __restrict__ in, _Float16* __restrict__ out, int n) {
  int i = blockIdx.x * blockDim.x + threadIdx.x;
  const int stride = gridDim.x * blockDim.x;
  for (; i < n; i += stride) out[i] = (_Float16)in[i];
}

// ---------------------------------------------------------------------------
// mask[b,t] = any(x[b,t,:] == -1.0); one wave per (b,t) row.
// ---------------------------------------------------------------------------
__global__ void k_mask(const float* __restrict__ x, float* __restrict__ mask, int rows) {
  const int row  = blockIdx.x * 8 + (threadIdx.x >> 5);
  const int lane = threadIdx.x & 31;
  if (row >= rows) return;
  const float* p = x + (size_t)row * 512;
  int bad = 0;
  for (int i = lane; i < 512; i += 32) bad |= (p[i] == -1.0f) ? 1 : 0;
  for (int off = 1; off < 32; off <<= 1) bad |= __shfl_xor(bad, off, 32);
  if (lane == 0) mask[row] = bad ? 1.0f : 0.0f;
}

// ---------------------------------------------------------------------------
// Generic C = A @ B^T + bias (f16 in, fp32 accum, f16 out).
// Block: 256 threads = 8 waves; each wave -> 16 rows x 64 cols; grid = (N/64, M/128).
// Depth-2 software pipeline: next K-chunk's 5 fragments are loaded into
// separate registers before the current 4 WMMAs issue, so the compiler can
// emit partial s_wait_loadcnt and keep loads in flight under the WMMAs.
// ---------------------------------------------------------------------------
__global__ __launch_bounds__(256) void k_gemm_nt(
    const _Float16* __restrict__ A, int lda,
    const _Float16* __restrict__ B, int ldb,
    const float* __restrict__ bias,
    _Float16* __restrict__ C, int ldc,
    int M, int K) {
  const int wave = threadIdx.x >> 5;
  const int m0   = blockIdx.y * 128 + wave * 16;
  const int n0   = blockIdx.x * 64;
  if (m0 >= M) return;

  const _Float16* Ap = A + (size_t)m0 * lda;

  v8f acc[4] = {};
  v16h a = load_fragA(Ap, lda);
  v16h b[4];
#pragma unroll
  for (int j = 0; j < 4; ++j) b[j] = load_fragB(B + (size_t)(n0 + j * 16) * ldb, ldb);

  for (int k = 0; k < K; k += 32) {
    const int kn = (k + 32 < K) ? (k + 32) : 0;   // clamped prefetch (scalar cond)
    v16h a2 = load_fragA(Ap + kn, lda);
    v16h b2[4];
#pragma unroll
    for (int j = 0; j < 4; ++j) b2[j] = load_fragB(B + (size_t)(n0 + j * 16) * ldb + kn, ldb);
#pragma unroll
    for (int j = 0; j < 4; ++j) acc[j] = wmma_f16(a, b[j], acc[j]);
    a = a2;
#pragma unroll
    for (int j = 0; j < 4; ++j) b[j] = b2[j];
  }

  const int lane = threadIdx.x & 31;
  const int colL = lane & 15;
  const int mb   = (lane >> 4) << 3;
#pragma unroll
  for (int j = 0; j < 4; ++j) {
    const int col = n0 + j * 16 + colL;
    const float bv = bias ? bias[col] : 0.0f;
#pragma unroll
    for (int v = 0; v < 8; ++v)
      C[(size_t)(m0 + mb + v) * ldc + col] = (_Float16)(acc[j][v] + bv);
  }
}

// ---------------------------------------------------------------------------
// Persistent GRU scan. Grid = 16 blocks; block owns 16 batch rows; h kept in LDS.
// Per step, wave w computes a 64-column slice of the recurrent GEMMs for all
// three gates (f16 WMMA, A-fragments from LDS), fuses sigmoid/tanh and update.
// Xz/Xr/Xh hold x@Wx^T + b (precomputed).  Wz/Wr/Wh point at full [512x1024]
// f16 weights; the recurrent part is columns 512..1023.
// global_prefetch of next step's gate inputs hides the per-step global reads.
// ---------------------------------------------------------------------------
__global__ __launch_bounds__(256) void k_gru_scan(
    const _Float16* __restrict__ Wz, const _Float16* __restrict__ Wr,
    const _Float16* __restrict__ Wh,
    const _Float16* __restrict__ Xz, const _Float16* __restrict__ Xr,
    const _Float16* __restrict__ Xh,
    const float* __restrict__ mask,
    _Float16* __restrict__ hOut, int T) {
  __shared__ _Float16 sH [16 * 512];  // h (f16 mirror, WMMA A operand)
  __shared__ _Float16 sRH[16 * 512];  // r*h (f16)
  __shared__ float    sHf[16 * 512];  // h master (f32)

  const int b0   = blockIdx.x * 16;
  const int wave = threadIdx.x >> 5;
  const int lane = threadIdx.x & 31;
  const int n0   = wave * 64;
  const int colL = lane & 15;
  const int mb   = (lane >> 4) << 3;

  for (int i = threadIdx.x; i < 16 * 512; i += 256) { sH[i] = (_Float16)0.0f; sHf[i] = 0.0f; }
  __syncthreads();

  for (int t = 0; t < T; ++t) {
    // prefetch next timestep's gate-input cachelines (gfx1250 global_prefetch)
    if (t + 1 < T) {
#pragma unroll
      for (int j = 0; j < 4; ++j) {
        const int col = n0 + j * 16 + colL;
        const size_t gn = ((size_t)(b0 + mb) * T + (t + 1)) * 512 + col;
        __builtin_prefetch(Xz + gn, 0, 1);
        __builtin_prefetch(Xr + gn, 0, 1);
        __builtin_prefetch(Xh + gn, 0, 1);
      }
    }
    // ---- z and r recurrent pre-activations: h @ U^T -------------------------
    v8f za[4] = {};
    v8f ra[4] = {};
    for (int k = 0; k < 512; k += 32) {
      v16h a = load_fragA(sH + k, 512);
#pragma unroll
      for (int j = 0; j < 4; ++j) {
        v16h bz = load_fragB(Wz + (size_t)(n0 + j * 16) * 1024 + 512 + k, 1024);
        za[j] = wmma_f16(a, bz, za[j]);
        v16h br = load_fragB(Wr + (size_t)(n0 + j * 16) * 1024 + 512 + k, 1024);
        ra[j] = wmma_f16(a, br, ra[j]);
      }
    }
    // ---- gates + r*h into LDS ----------------------------------------------
    float zreg[4][8];
#pragma unroll
    for (int j = 0; j < 4; ++j) {
      const int col = n0 + j * 16 + colL;
#pragma unroll
      for (int v = 0; v < 8; ++v) {
        const int row = mb + v;
        const size_t g = ((size_t)(b0 + row) * T + t) * 512 + col;
        const float zp = za[j][v] + (float)Xz[g];
        const float rp = ra[j][v] + (float)Xr[g];
        float zv = 1.0f / (1.0f + __expf(-zp));
        if (mask[(size_t)(b0 + row) * T + t] != 0.0f) zv = 1.0f;  // masked: carry h
        const float rv = 1.0f / (1.0f + __expf(-rp));
        zreg[j][v] = zv;
        sRH[row * 512 + col] = (_Float16)(rv * sHf[row * 512 + col]);
      }
    }
    __syncthreads();
    // ---- candidate: (r*h) @ Uh^T -------------------------------------------
    v8f ha[4] = {};
    for (int k = 0; k < 512; k += 32) {
      v16h a = load_fragA(sRH + k, 512);
#pragma unroll
      for (int j = 0; j < 4; ++j) {
        v16h bh = load_fragB(Wh + (size_t)(n0 + j * 16) * 1024 + 512 + k, 1024);
        ha[j] = wmma_f16(a, bh, ha[j]);
      }
    }
    // ---- update h (each wave owns its 64 columns: no cross-wave conflict) ---
#pragma unroll
    for (int j = 0; j < 4; ++j) {
      const int col = n0 + j * 16 + colL;
#pragma unroll
      for (int v = 0; v < 8; ++v) {
        const int row = mb + v;
        const size_t g = ((size_t)(b0 + row) * T + t) * 512 + col;
        const float hp = ha[j][v] + (float)Xh[g];
        const float e  = __expf(-2.0f * hp);
        const float th = (1.0f - e) / (1.0f + e);      // tanh
        const float ho = sHf[row * 512 + col];
        const float zv = zreg[j][v];
        const float hn = zv * ho + (1.0f - zv) * th;
        sHf[row * 512 + col] = hn;
        sH [row * 512 + col] = (_Float16)hn;
      }
    }
    __syncthreads();
  }

  for (int i = threadIdx.x; i < 16 * 512; i += 256)
    hOut[(size_t)b0 * 512 + i] = sH[i];
}

// ---------------------------------------------------------------------------
// Attention at t = T-1 over the batch axis. One block per head (8 blocks).
// Each wave: 2 row-blocks of 16 queries; scores 16x256 live entirely in 16
// accumulator v8f's; softmax is register-resident with __shfl_xor reductions
// (C-layout: row m = VGPR slot m&7 across one 16-lane half).
// Dynamic LDS: Vt[64][256] f16 + per-wave probs[16][256] f16 + mask[256] f32.
// ---------------------------------------------------------------------------
__global__ __launch_bounds__(256) void k_attn(
    const _Float16* __restrict__ qkv,   // [256 x 1536]
    const float* __restrict__ mask,     // [B*T]
    _Float16* __restrict__ ctx,         // [256 x 512]
    int T) {
  extern __shared__ unsigned char smem[];
  _Float16* sVt   = (_Float16*)smem;                              // 64*256
  _Float16* sP    = (_Float16*)(smem + 64 * 256 * 2);             // 8*16*256
  float*    sMask = (float*)(smem + 64 * 256 * 2 + 8 * 16 * 256 * 2);

  const int head = blockIdx.x;
  const int wave = threadIdx.x >> 5;
  const int lane = threadIdx.x & 31;
  const int colL = lane & 15;
  const int mb   = (lane >> 4) << 3;

  // V^T into LDS; key mask for last timestep.
  for (int i = threadIdx.x; i < 64 * 256; i += 256) {
    const int c = i >> 8, r = i & 255;
    sVt[i] = qkv[(size_t)r * 1536 + 1024 + head * 64 + c];
  }
  if (threadIdx.x < 256) sMask[threadIdx.x] = mask[(size_t)threadIdx.x * T + (T - 1)];
  __syncthreads();

  const _Float16* Qb = qkv + head * 64;
  const _Float16* Kb = qkv + 512 + head * 64;
  _Float16* myP = sP + wave * 16 * 256;

  for (int it = 0; it < 2; ++it) {
    const int m0 = (wave * 2 + it) * 16;

    // scores = Q @ K^T (hd = 64)
    v8f acc[16] = {};
    for (int k = 0; k < 64; k += 32) {
      v16h a = load_fragA(Qb + (size_t)m0 * 1536 + k, 1536);
#pragma unroll
      for (int j = 0; j < 16; ++j) {
        v16h b = load_fragB(Kb + (size_t)(j * 16) * 1536 + k, 1536);
        acc[j] = wmma_f16(a, b, acc[j]);
      }
    }
    // scale + key mask
#pragma unroll
    for (int j = 0; j < 16; ++j) {
      const float mv = sMask[j * 16 + colL];
#pragma unroll
      for (int v = 0; v < 8; ++v) {
        const float s = acc[j][v] * 0.125f;             // 1/sqrt(64)
        acc[j][v] = (mv != 0.0f) ? -1e30f : s;
      }
    }
    // register softmax: slot v holds row mb+v for this half-wave
#pragma unroll
    for (int v = 0; v < 8; ++v) {
      float m = -1e30f;
#pragma unroll
      for (int j = 0; j < 16; ++j) m = fmaxf(m, acc[j][v]);
      for (int off = 1; off < 16; off <<= 1) m = fmaxf(m, __shfl_xor(m, off, 32));
      float s = 0.0f;
#pragma unroll
      for (int j = 0; j < 16; ++j) { const float e = __expf(acc[j][v] - m); acc[j][v] = e; s += e; }
      for (int off = 1; off < 16; off <<= 1) s += __shfl_xor(s, off, 32);
      const float rinv = 1.0f / s;
#pragma unroll
      for (int j = 0; j < 16; ++j)
        myP[(mb + v) * 256 + j * 16 + colL] = (_Float16)(acc[j][v] * rinv);
    }
    // ctx = P @ V  ([16x256] @ [256x64]) from LDS
    v8f cacc[4] = {};
    for (int k = 0; k < 256; k += 32) {
      v16h a = load_fragA(myP + k, 256);
#pragma unroll
      for (int j = 0; j < 4; ++j) {
        v16h b = load_fragB(sVt + (size_t)(j * 16) * 256 + k, 256);
        cacc[j] = wmma_f16(a, b, cacc[j]);
      }
    }
#pragma unroll
    for (int j = 0; j < 4; ++j)
#pragma unroll
      for (int v = 0; v < 8; ++v)
        ctx[(size_t)(m0 + mb + v) * 512 + head * 64 + j * 16 + colL] = (_Float16)cacc[j][v];
  }
}

// ---------------------------------------------------------------------------
// Final FC (H=512 -> C=2) + sigmoid, fp32.
// ---------------------------------------------------------------------------
__global__ void k_fc(const _Float16* __restrict__ attn, const float* __restrict__ fcw,
                     const float* __restrict__ fcb, float* __restrict__ out) {
  const int b = blockIdx.x * blockDim.x + threadIdx.x;
  if (b >= 256) return;
#pragma unroll
  for (int c = 0; c < 2; ++c) {
    float s = fcb[c];
    for (int k = 0; k < 512; ++k) s += (float)attn[b * 512 + k] * fcw[c * 512 + k];
    out[b * 2 + c] = 1.0f / (1.0f + __expf(-s));
  }
}

// ---------------------------------------------------------------------------
extern "C" void kernel_launch(void* const* d_in, const int* in_sizes, int n_in,
                              void* d_out, int out_size, void* d_ws, size_t ws_size,
                              hipStream_t stream) {
  (void)in_sizes; (void)n_in; (void)out_size; (void)ws_size;
  const float* x   = (const float*)d_in[0];
  const float* Wz  = (const float*)d_in[1];
  const float* bz  = (const float*)d_in[2];
  const float* Wr  = (const float*)d_in[3];
  const float* br  = (const float*)d_in[4];
  const float* Wh  = (const float*)d_in[5];
  const float* bh  = (const float*)d_in[6];
  const float* ipw = (const float*)d_in[7];
  const float* ipb = (const float*)d_in[8];
  const float* opw = (const float*)d_in[9];
  const float* opb = (const float*)d_in[10];
  const float* fcw = (const float*)d_in[11];
  const float* fcb = (const float*)d_in[12];
  float* out = (float*)d_out;

  const int B = 256, T = 128, I = 512, H = 512;
  const size_t BT = (size_t)B * T;

  size_t off = 0;
  auto carve = [&](size_t bytes) -> void* {
    void* p = (char*)d_ws + off;
    off += (bytes + 255) & ~(size_t)255;
    return p;
  };
  _Float16* x16  = (_Float16*)carve(BT * I * 2);            // x in f16
  _Float16* Wz16 = (_Float16*)carve((size_t)H * (I + H) * 2);
  _Float16* Wr16 = (_Float16*)carve((size_t)H * (I + H) * 2);
  _Float16* Wh16 = (_Float16*)carve((size_t)H * (I + H) * 2);
  _Float16* ip16 = (_Float16*)carve((size_t)3 * H * H * 2); // in_proj [1536 x 512]
  _Float16* op16 = (_Float16*)carve((size_t)H * H * 2);     // out_proj [512 x 512]
  _Float16* Xz   = (_Float16*)carve(BT * H * 2);            // x@Wx^T + b per gate
  _Float16* Xr   = (_Float16*)carve(BT * H * 2);
  _Float16* Xh   = (_Float16*)carve(BT * H * 2);
  float*    mskf = (float*)carve(BT * 4);
  _Float16* h16  = (_Float16*)carve((size_t)B * H * 2);     // final hidden state
  _Float16* qkv  = (_Float16*)carve((size_t)B * 3 * H * 2);
  _Float16* ctx  = (_Float16*)carve((size_t)B * H * 2);
  _Float16* att  = (_Float16*)carve((size_t)B * H * 2);

  // 1) precision conversion
  k_f32_to_f16<<<4096, 256, 0, stream>>>(x, x16, (int)(BT * I));
  k_f32_to_f16<<<512, 256, 0, stream>>>(Wz, Wz16, H * (I + H));
  k_f32_to_f16<<<512, 256, 0, stream>>>(Wr, Wr16, H * (I + H));
  k_f32_to_f16<<<512, 256, 0, stream>>>(Wh, Wh16, H * (I + H));
  k_f32_to_f16<<<512, 256, 0, stream>>>(ipw, ip16, 3 * H * H);
  k_f32_to_f16<<<512, 256, 0, stream>>>(opw, op16, H * H);

  // 2) padding mask
  k_mask<<<(int)(BT / 8), 256, 0, stream>>>(x, mskf, (int)BT);

  // 3) hoisted x-part of all gate pre-activations: [BT x I] @ Wx^T + b
  {
    dim3 grid(H / 64, (unsigned)(BT / 128));
    k_gemm_nt<<<grid, 256, 0, stream>>>(x16, I, Wz16, I + H, bz, Xz, H, (int)BT, I);
    k_gemm_nt<<<grid, 256, 0, stream>>>(x16, I, Wr16, I + H, br, Xr, H, (int)BT, I);
    k_gemm_nt<<<grid, 256, 0, stream>>>(x16, I, Wh16, I + H, bh, Xh, H, (int)BT, I);
  }

  // 4) sequential GRU scan (persistent, h in LDS)
  k_gru_scan<<<B / 16, 256, 0, stream>>>(Wz16, Wr16, Wh16, Xz, Xr, Xh, mskf, h16, T);

  // 5) qkv projection on final hidden state: [256 x 512] @ [1536 x 512]^T
  {
    dim3 grid(3 * H / 64, B / 128);
    k_gemm_nt<<<grid, 256, 0, stream>>>(h16, H, ip16, H, ipb, qkv, 3 * H, B, H);
  }

  // 6) per-head attention over batch at t = T-1
  {
    const size_t shmem = (size_t)64 * 256 * 2 + (size_t)8 * 16 * 256 * 2 + 256 * 4;
    k_attn<<<8, 256, shmem, stream>>>(qkv, mskf, ctx, T);
  }

  // 7) out_proj: [256 x 512] @ [512 x 512]^T
  {
    dim3 grid(H / 64, B / 128);
    k_gemm_nt<<<grid, 256, 0, stream>>>(ctx, H, op16, H, opb, att, H, B, H);
  }

  // 8) fc + sigmoid -> d_out [256,1,2] f32
  k_fc<<<1, 256, 0, stream>>>(att, fcw, fcb, out);
}
